// MultiHeadAttention_65678639890537
// MI455X (gfx1250) — matrix operand order repaired and checked by reading, not compiled, not run on hardware
//
#include <hip/hip_runtime.h>
#include <hip/hip_bf16.h>

// ---------------------------------------------------------------------------
// MultiHeadAttention for MI455X (gfx1250): bf16 WMMA everywhere,
// TDM (tensor_load_to_lds) + async global->LDS staging.
// B=4, S=2048, D=1024, H=16, DH=64.
// ---------------------------------------------------------------------------

typedef __bf16 bf16_t;
typedef bf16_t v16bf __attribute__((ext_vector_type(16)));
typedef float  v8f   __attribute__((ext_vector_type(8)));
typedef float  f32x4 __attribute__((ext_vector_type(4)));
typedef unsigned short u16x8 __attribute__((ext_vector_type(8)));
typedef unsigned int   u32x4v __attribute__((ext_vector_type(4)));
typedef unsigned int   u32x8v __attribute__((ext_vector_type(8)));

union WFrag { v16bf v; u16x8 h[2]; };

__device__ __forceinline__ unsigned short f2bf(float x) {
  union { float f; unsigned u; } c; c.f = x;
  unsigned r = c.u + 0x7FFFu + ((c.u >> 16) & 1u);   // round-to-nearest-even
  return (unsigned short)(r >> 16);
}

// Async global->LDS 16-byte copy (CDNA5 GLOBAL_LOAD_ASYNC_TO_LDS_B128, GV mode).
// Tracked by ASYNCcnt; caller must wait_async() before consuming the LDS data.
__device__ __forceinline__ void async_g2l_b128(const void* gptr, unsigned lds_addr) {
  asm volatile("global_load_async_to_lds_b128 %0, %1, off"
               :: "v"(lds_addr), "v"(gptr) : "memory");
}
__device__ __forceinline__ void wait_async() {
  asm volatile("s_wait_asynccnt 0" ::: "memory");
}
__device__ __forceinline__ unsigned lds_addr_of(const void* p) {
  // Generic shared pointers carry the LDS byte offset in addr[31:0].
  return (unsigned)(unsigned long long)p;
}

// ---------------------------------------------------------------------------
// Tensor Data Mover: 2D bf16 tile, global -> LDS, with LDS row padding.
// D# per CDNA5 ISA:
//   group0: [1:0]=count=1, [63:32]=lds_addr, [120:64]=global_addr, [127:126]=2
//   group1: data_size=2B, pad_enable, pad_interval code 4 (32 DWORDs = 128B),
//           pad_amount code 3 (4 DWORDs = 16B) -> 144B LDS row stride;
//           tensor_dim0 = tile_dim0 = 64, tile_dim1 = 64, stride0 = 64.
// Tracked by TENSORcnt; wait with s_wait_tensorcnt before the barrier.
// ---------------------------------------------------------------------------
__device__ __forceinline__ void tdm_load_tile_64x64_bf16(const void* gbase,
                                                         unsigned lds_base,
                                                         unsigned tensor_rows) {
  const unsigned long long ga = (unsigned long long)gbase;
  u32x4v g0;
  g0[0] = 1u;                                            // count=1 (valid, user)
  g0[1] = lds_base;                                      // LDS byte address
  g0[2] = (unsigned)ga;                                  // global_addr[31:0]
  g0[3] = (unsigned)((ga >> 32) & 0x01FFFFFFu)           // global_addr[56:32]
        | 0x80000000u;                                   // type = 2 ("image")
  u32x8v g1;
  g1[0] = (1u << 16)                                     // data_size = 2 bytes
        | (1u << 20)                                     // pad_enable
        | (4u << 22)                                     // pad_interval: 32 DW
        | (3u << 25);                                    // pad_amount: 4 DW
  g1[1] = (64u & 0xFFFFu) << 16;                         // tensor_dim0 lo16
  g1[2] = (64u >> 16) | ((tensor_rows & 0xFFFFu) << 16); // dim0 hi | dim1 lo
  g1[3] = (tensor_rows >> 16) | (64u << 16);             // dim1 hi | tile_dim0
  g1[4] = 64u;                                           // tile_dim1 (dim2 = 0)
  g1[5] = 64u;                                           // tensor_dim0_stride lo32
  g1[6] = 0u;                                            // stride0 hi | stride1 lo
  g1[7] = 0u;                                            // stride1 hi
  asm volatile("tensor_load_to_lds %0, %1" :: "s"(g0), "s"(g1) : "memory");
}

#define S_LEN 2048
#define D_LEN 1024
#define H_NUM 16

// ---------------------------------------------------------------------------
// Kernel 1: fused Q/K/V projection.  C[m,n] = sum_k X[m,k]*W[n,k] + b[n]
// 128x128 tile / 256 threads (8 waves). blockIdx.z picks the projection.
// Output bf16 in [B,H,S,DH] layout; Q pre-scaled by 1/sqrt(DH)=0.125.
// (fp32 -> bf16 conversion is fused into staging; async path not usable here.)
// ---------------------------------------------------------------------------
__global__ __launch_bounds__(256)
void mha_proj_gemm_65678(const float* __restrict__ xq,
                         const float* __restrict__ xk,
                         const float* __restrict__ xv,
                         const float* __restrict__ Wq, const float* __restrict__ bq,
                         const float* __restrict__ Wk, const float* __restrict__ bk,
                         const float* __restrict__ Wv, const float* __restrict__ bv,
                         unsigned short* __restrict__ qo,
                         unsigned short* __restrict__ ko,
                         unsigned short* __restrict__ vo)
{
  const int z = blockIdx.z;
  const float* A    = (z == 0) ? xq : (z == 1) ? xk : xv;
  const float* W    = (z == 0) ? Wq : (z == 1) ? Wk : Wv;
  const float* bias = (z == 0) ? bq : (z == 1) ? bk : bv;
  unsigned short* out = (z == 0) ? qo : (z == 1) ? ko : vo;
  const float scale = (z == 0) ? 0.125f : 1.0f;

  __shared__ __align__(16) unsigned short As[128][40];  // 80B stride (16B mult)
  __shared__ __align__(16) unsigned short Ws[128][40];

  const int tid  = threadIdx.x;
  const int lane = tid & 31, wave = tid >> 5;
  const int hlf  = lane >> 4, l15 = lane & 15;
  const int waveM = wave & 3, waveN = wave >> 2;        // 4 x 2 wave grid
  const long mBase = (long)blockIdx.y * 128;
  const int  nBase = blockIdx.x * 128;

  v8f acc[2][4];
  #pragma unroll
  for (int mt = 0; mt < 2; ++mt)
    #pragma unroll
    for (int nt = 0; nt < 4; ++nt)
      acc[mt][nt] = (v8f){};

  const int r0 = tid >> 2;          // 0..63
  const int c0 = (tid & 3) * 8;     // 0,8,16,24

  for (int kk = 0; kk < 32; ++kk) {
    const int k0 = kk * 32;
    // Stage A and W tiles (fp32 -> bf16) into LDS.
    #pragma unroll
    for (int rr = 0; rr < 128; rr += 64) {
      const int row = r0 + rr;
      const float* ap = A + (mBase + row) * D_LEN + k0 + c0;
      const float* wp = W + (long)(nBase + row) * D_LEN + k0 + c0;
      f32x4 a0 = *(const f32x4*)ap;
      f32x4 a1 = *(const f32x4*)(ap + 4);
      f32x4 w0 = *(const f32x4*)wp;
      f32x4 w1 = *(const f32x4*)(wp + 4);
      u16x8 ah, wh;
      #pragma unroll
      for (int j = 0; j < 4; ++j) {
        ah[j] = f2bf(a0[j]); ah[j + 4] = f2bf(a1[j]);
        wh[j] = f2bf(w0[j]); wh[j + 4] = f2bf(w1[j]);
      }
      *(u16x8*)&As[row][c0] = ah;
      *(u16x8*)&Ws[row][c0] = wh;
    }
    __syncthreads();

    if (kk + 1 < 32) {   // global_prefetch_b8 next K slice
      __builtin_prefetch(A + (mBase + r0) * D_LEN + k0 + 32 + c0, 0, 1);
      __builtin_prefetch(W + (long)(nBase + r0) * D_LEN + k0 + 32 + c0, 0, 1);
    }

    // A fragments: lane<16 holds M=l15, K chunks {hlf*8, 16+hlf*8}.
    WFrag af[2];
    #pragma unroll
    for (int mt = 0; mt < 2; ++mt) {
      const int row = waveM * 32 + mt * 16 + l15;
      af[mt].h[0] = *(const u16x8*)&As[row][hlf * 8];
      af[mt].h[1] = *(const u16x8*)&As[row][16 + hlf * 8];
    }
    // B fragments: lane holds column N=l15, K chunks {hlf*8, 16+hlf*8}.
    #pragma unroll
    for (int nt = 0; nt < 4; ++nt) {
      const int rn = waveN * 64 + nt * 16 + l15;
      WFrag bfr;
      bfr.h[0] = *(const u16x8*)&Ws[rn][hlf * 8];
      bfr.h[1] = *(const u16x8*)&Ws[rn][16 + hlf * 8];
      #pragma unroll
      for (int mt = 0; mt < 2; ++mt)
        acc[mt][nt] = __builtin_amdgcn_wmma_f32_16x16x32_bf16(
            false, af[mt].v, false, bfr.v, (short)0, acc[mt][nt], false, false);
    }
    __syncthreads();
  }

  // Epilogue: bias, scale, scatter to [B,H,S,DH] as bf16.
  #pragma unroll
  for (int nt = 0; nt < 4; ++nt) {
    const int n = nBase + waveN * 64 + nt * 16 + l15;
    const float bb = bias[n];
    const int h = n >> 6, dh = n & 63;
    #pragma unroll
    for (int mt = 0; mt < 2; ++mt) {
      #pragma unroll
      for (int r = 0; r < 8; ++r) {
        const long m = mBase + waveM * 32 + mt * 16 + r + hlf * 8;
        const long bi = m >> 11, s = m & (S_LEN - 1);
        out[(((bi * H_NUM + h) * S_LEN + s) << 6) + dh] =
            f2bf((acc[mt][nt][r] + bb) * scale);
      }
    }
  }
}

// ---------------------------------------------------------------------------
// Kernel 2: causal flash attention. One block = (b, h, 64-row q tile),
// 128 threads = 4 waves, each wave owns 16 query rows.
// K tile staged by the Tensor Data Mover; V staged via VGPRs (transposed).
// ---------------------------------------------------------------------------
__global__ __launch_bounds__(128)
void mha_flash_65678(const unsigned short* __restrict__ Qb,
                     const unsigned short* __restrict__ Kb,
                     const unsigned short* __restrict__ Vb,
                     unsigned short* __restrict__ Ctx)
{
  __shared__ __align__(16) unsigned short Ks[64][72];  // [kv][dh]
  __shared__ __align__(16) unsigned short Vt[64][72];  // [dh][kv] (transposed)
  __shared__ __align__(16) unsigned short Pt[64][72];  // [q][kv]

  const int tid  = threadIdx.x;
  const int lane = tid & 31, wave = tid >> 5;
  const int hlf  = lane >> 4, l15 = lane & 15;
  const int qt = blockIdx.x, hh = blockIdx.y, bb = blockIdx.z;
  const int qBase = qt * 64;
  const long headBase = ((long)(bb * H_NUM + hh)) * S_LEN;

  // Q fragments: persistent for the whole kv loop. Q already scaled by 0.125.
  const unsigned short* qrow = Qb + ((headBase + qBase + wave * 16 + l15) << 6);
  WFrag qf[2];
  #pragma unroll
  for (int ks = 0; ks < 2; ++ks) {
    qf[ks].h[0] = *(const u16x8*)(qrow + ks * 32 + hlf * 8);
    qf[ks].h[1] = *(const u16x8*)(qrow + ks * 32 + 16 + hlf * 8);
  }

  float m_st[8], l_st[8];
  v8f o[4] = {};
  #pragma unroll
  for (int r = 0; r < 8; ++r) { m_st[r] = -1e30f; l_st[r] = 0.0f; }

  for (int kt = 0; kt <= qt; ++kt) {
    // Stage K via TDM (one descriptor; EXEC-independent wave-level DMA)
    // and V transposed via VGPRs.
    const unsigned short* Kg = Kb + ((headBase + kt * 64) << 6);
    const unsigned short* Vg = Vb + ((headBase + kt * 64) << 6);
    if (wave == 0) {
      tdm_load_tile_64x64_bf16(Kg, lds_addr_of(&Ks[0][0]),
                               (unsigned)(S_LEN - kt * 64));
    }
    #pragma unroll
    for (int i = 0; i < 4; ++i) {
      const int c   = tid + 128 * i;     // 512 chunks of 8 bf16
      const int row = c >> 3;            // kv row 0..63
      const int cg  = (c & 7) * 8;       // dh base
      const u16x8 vv = *(const u16x8*)(Vg + row * 64 + cg);
      #pragma unroll
      for (int j = 0; j < 8; ++j) Vt[cg + j][row] = vv[j];
    }
    __builtin_amdgcn_s_wait_tensorcnt(0);
    __syncthreads();

    // S = Q * K^T  (16x64 per wave; 2 K-steps x 4 N-tiles of WMMA).
    v8f sa[4] = {};
    #pragma unroll
    for (int ks = 0; ks < 2; ++ks) {
      #pragma unroll
      for (int nt = 0; nt < 4; ++nt) {
        const int rn = nt * 16 + l15;          // kv row = B column
        WFrag bfr;
        bfr.h[0] = *(const u16x8*)&Ks[rn][ks * 32 + hlf * 8];
        bfr.h[1] = *(const u16x8*)&Ks[rn][ks * 32 + 16 + hlf * 8];
        sa[nt] = __builtin_amdgcn_wmma_f32_16x16x32_bf16(
            false, qf[ks].v, false, bfr.v, (short)0, sa[nt], false, false);
      }
    }

    // Causal mask (only the diagonal tile needs it).
    if (kt == qt) {
      #pragma unroll
      for (int nt = 0; nt < 4; ++nt) {
        const int kg = kt * 64 + nt * 16 + l15;
        #pragma unroll
        for (int r = 0; r < 8; ++r) {
          const int qg = qBase + wave * 16 + r + hlf * 8;
          if (kg > qg) sa[nt][r] = -1e30f;
        }
      }
    }

    // Online softmax: half-wave (16 lane) reductions per row.
    float al[8];
    #pragma unroll
    for (int r = 0; r < 8; ++r) {
      float mx = fmaxf(fmaxf(sa[0][r], sa[1][r]), fmaxf(sa[2][r], sa[3][r]));
      mx = fmaxf(mx, __shfl_xor(mx, 1));
      mx = fmaxf(mx, __shfl_xor(mx, 2));
      mx = fmaxf(mx, __shfl_xor(mx, 4));
      mx = fmaxf(mx, __shfl_xor(mx, 8));
      const float mn = fmaxf(m_st[r], mx);
      al[r] = __expf(m_st[r] - mn);
      m_st[r] = mn;
    }
    #pragma unroll
    for (int r = 0; r < 8; ++r) {
      float rs = 0.0f;
      #pragma unroll
      for (int nt = 0; nt < 4; ++nt) {
        const float p = __expf(sa[nt][r] - m_st[r]);
        rs += p;
        Pt[wave * 16 + r + hlf * 8][nt * 16 + l15] = f2bf(p);
      }
      rs += __shfl_xor(rs, 1);
      rs += __shfl_xor(rs, 2);
      rs += __shfl_xor(rs, 4);
      rs += __shfl_xor(rs, 8);
      l_st[r] = l_st[r] * al[r] + rs;
      #pragma unroll
      for (int dt = 0; dt < 4; ++dt) o[dt][r] *= al[r];
    }

    // O += P * V : A from Pt (same-wave rows), B from transposed Vt.
    #pragma unroll
    for (int ks = 0; ks < 2; ++ks) {
      WFrag pf;
      const int prow = wave * 16 + l15;
      pf.h[0] = *(const u16x8*)&Pt[prow][ks * 32 + hlf * 8];
      pf.h[1] = *(const u16x8*)&Pt[prow][ks * 32 + 16 + hlf * 8];
      #pragma unroll
      for (int dt = 0; dt < 4; ++dt) {
        const int rd = dt * 16 + l15;          // dh = B column
        WFrag bfr;
        bfr.h[0] = *(const u16x8*)&Vt[rd][ks * 32 + hlf * 8];
        bfr.h[1] = *(const u16x8*)&Vt[rd][ks * 32 + 16 + hlf * 8];
        o[dt] = __builtin_amdgcn_wmma_f32_16x16x32_bf16(
            false, pf.v, false, bfr.v, (short)0, o[dt], false, false);
      }
    }
    __syncthreads();   // before K/V tiles are overwritten
  }

  // Normalize and write context, bf16 [B,S,D] layout.
  #pragma unroll
  for (int r = 0; r < 8; ++r) {
    const float inv = 1.0f / l_st[r];
    const int s = qBase + wave * 16 + r + hlf * 8;
    const long base = ((long)bb * S_LEN + s) * D_LEN + hh * 64;
    #pragma unroll
    for (int dt = 0; dt < 4; ++dt)
      Ctx[base + dt * 16 + l15] = f2bf(o[dt][r] * inv);
  }
}

// ---------------------------------------------------------------------------
// Kernel 3: output projection. A is bf16 context (async-staged);
// output fp32 [B,S,D].
// ---------------------------------------------------------------------------
__global__ __launch_bounds__(256)
void mha_out_gemm_65678(const unsigned short* __restrict__ Actx,
                        const float* __restrict__ Wo,
                        const float* __restrict__ bo,
                        float* __restrict__ out)
{
  __shared__ __align__(16) unsigned short As[128][40];
  __shared__ __align__(16) unsigned short Ws[128][40];

  const int tid  = threadIdx.x;
  const int lane = tid & 31, wave = tid >> 5;
  const int hlf  = lane >> 4, l15 = lane & 15;
  const int waveM = wave & 3, waveN = wave >> 2;
  const long mBase = (long)blockIdx.y * 128;
  const int  nBase = blockIdx.x * 128;

  v8f acc[2][4];
  #pragma unroll
  for (int mt = 0; mt < 2; ++mt)
    #pragma unroll
    for (int nt = 0; nt < 4; ++nt)
      acc[mt][nt] = (v8f){};

  const int r0 = tid >> 2;
  const int c0 = (tid & 3) * 8;

  for (int kk = 0; kk < 32; ++kk) {
    const int k0 = kk * 32;
    #pragma unroll
    for (int rr = 0; rr < 128; rr += 64) {
      const int row = r0 + rr;
      // Context tile is already bf16: async copy straight into LDS.
      async_g2l_b128(Actx + (mBase + row) * D_LEN + k0 + c0,
                     lds_addr_of(&As[row][c0]));
      const float* wp = Wo + (long)(nBase + row) * D_LEN + k0 + c0;
      f32x4 w0 = *(const f32x4*)wp;
      f32x4 w1 = *(const f32x4*)(wp + 4);
      u16x8 wh;
      #pragma unroll
      for (int j = 0; j < 4; ++j) { wh[j] = f2bf(w0[j]); wh[j + 4] = f2bf(w1[j]); }
      *(u16x8*)&Ws[row][c0] = wh;
    }
    wait_async();
    __syncthreads();

    if (kk + 1 < 32) {
      __builtin_prefetch(Actx + (mBase + r0) * D_LEN + k0 + 32 + c0, 0, 1);
      __builtin_prefetch(Wo + (long)(nBase + r0) * D_LEN + k0 + 32 + c0, 0, 1);
    }

    WFrag af[2];
    #pragma unroll
    for (int mt = 0; mt < 2; ++mt) {
      const int row = waveM * 32 + mt * 16 + l15;
      af[mt].h[0] = *(const u16x8*)&As[row][hlf * 8];
      af[mt].h[1] = *(const u16x8*)&As[row][16 + hlf * 8];
    }
    #pragma unroll
    for (int nt = 0; nt < 4; ++nt) {
      const int rn = waveN * 64 + nt * 16 + l15;
      WFrag bfr;
      bfr.h[0] = *(const u16x8*)&Ws[rn][hlf * 8];
      bfr.h[1] = *(const u16x8*)&Ws[rn][16 + hlf * 8];
      #pragma unroll
      for (int mt = 0; mt < 2; ++mt)
        acc[mt][nt] = __builtin_amdgcn_wmma_f32_16x16x32_bf16(
            false, af[mt].v, false, bfr.v, (short)0, acc[mt][nt], false, false);
    }
    __syncthreads();
  }

  #pragma unroll
  for (int nt = 0; nt < 4; ++nt) {
    const int n = nBase + waveN * 64 + nt * 16 + l15;
    const float bbv = bo[n];
    #pragma unroll
    for (int mt = 0; mt < 2; ++mt) {
      #pragma unroll
      for (int r = 0; r < 8; ++r) {
        const long m = mBase + waveM * 32 + mt * 16 + r + hlf * 8;
        out[m * D_LEN + n] = acc[mt][nt][r] + bbv;
      }
    }
  }
}

// ---------------------------------------------------------------------------
extern "C" void kernel_launch(void* const* d_in, const int* in_sizes, int n_in,
                              void* d_out, int out_size, void* d_ws, size_t ws_size,
                              hipStream_t stream) {
  const float* q  = (const float*)d_in[0];
  const float* k  = (const float*)d_in[1];
  const float* v  = (const float*)d_in[2];
  // d_in[3] is the causal tril mask; it is implemented analytically.
  const float* Wq = (const float*)d_in[4];
  const float* bq = (const float*)d_in[5];
  const float* Wk = (const float*)d_in[6];
  const float* bk = (const float*)d_in[7];
  const float* Wv = (const float*)d_in[8];
  const float* bv = (const float*)d_in[9];
  const float* Wo = (const float*)d_in[10];
  const float* bo = (const float*)d_in[11];
  float* out = (float*)d_out;

  const long NEL = 4L * S_LEN * D_LEN;          // 8.4M elements per tensor
  unsigned short* qb = (unsigned short*)d_ws;   // bf16 Q [B,H,S,DH] (pre-scaled)
  unsigned short* kb = qb + NEL;                // bf16 K [B,H,S,DH]
  unsigned short* vb = kb + NEL;                // bf16 V [B,H,S,DH]
  unsigned short* cb = vb + NEL;                // bf16 context [B,S,D]

  mha_proj_gemm_65678<<<dim3(8, 64, 3), 256, 0, stream>>>(
      q, k, v, Wq, bq, Wk, bk, Wv, bv, qb, kb, vb);
  mha_flash_65678<<<dim3(32, 16, 4), 128, 0, stream>>>(qb, kb, vb, cb);
  mha_out_gemm_65678<<<dim3(8, 64, 1), 256, 0, stream>>>(cb, Wo, bo, out);
}
